// HSTU_66279935312625
// MI455X (gfx1250) — matrix operand order repaired and checked by reading, not compiled, not run on hardware
//
#include <hip/hip_runtime.h>
#include <hip/hip_bf16.h>

typedef __attribute__((ext_vector_type(16))) __bf16 bf16x16;
typedef __attribute__((ext_vector_type(8)))  __bf16 bf16x8;
typedef __attribute__((ext_vector_type(8)))  float  f32x8;
typedef __attribute__((ext_vector_type(4)))  unsigned int u32x4;
typedef __attribute__((ext_vector_type(8)))  int i32x8;
typedef __attribute__((ext_vector_type(4)))  int i32x4;

#define B_  4
#define L_  2048
#define D_  1024
#define H_  16
#define HD_ 64
#define M_  (B_ * L_)     // 8192 rows
#define N1_ (4 * D_)      // 4096 proj cols
#define KT  32            // k-step
#define LDT 40            // padded LDS row stride (bf16): 80B, 16B-aligned, bank-clean

static __device__ __forceinline__ float fast_sigmoid(float x) {
  return 1.0f / (1.0f + __expf(-x));
}

// --- CDNA5 async global->LDS copy (ASYNCcnt path), 16 bytes per lane ---------
static __device__ __forceinline__ void async_cp16(const void* g, void* lds) {
  unsigned loff = (unsigned)(size_t)lds;     // generic LDS addr low 32 bits = LDS offset
  asm volatile("global_load_async_to_lds_b128 %0, %1, off"
               :: "v"(loff), "v"(g) : "memory");
}
static __device__ __forceinline__ void async_wait0() {
  asm volatile("s_wait_asynccnt 0" ::: "memory");
}
static __device__ __forceinline__ void ds_wait0() {
  asm volatile("s_wait_dscnt 0" ::: "memory");
}
static __device__ __forceinline__ void tensor_wait0() {
  asm volatile("s_wait_tensorcnt 0" ::: "memory");
}

// --- CDNA5 Tensor Data Mover: 2D tile global->LDS, with HW LDS padding -------
// Copies tile_h rows x tile_w elems (2B each, rows strided by row_stride elems)
// into LDS with 16B padding inserted after every 64B row (net stride = LDT).
static __device__ __forceinline__ void tdm_load_tile_2d(
    const void* gaddr, void* lds, int tile_w, int tile_h,
    int tensor_w, int tensor_h, int row_stride) {
  unsigned glo = (unsigned)(size_t)gaddr;
  unsigned ghi = (unsigned)(((size_t)gaddr) >> 32);
  u32x4 g0;
  g0[0] = 1u;                                   // count=1, user descriptor
  g0[1] = (unsigned)(size_t)lds;                // lds_addr (bytes)
  g0[2] = glo;                                  // global_addr[31:0]
  g0[3] = (ghi & 0x01FFFFFFu) | (2u << 30);     // global_addr[56:32], type=2 (image)
  i32x8 g1;
  g1[0] = (1 << 16)     // data_size = 2B
        | (1 << 20)     // pad_enable
        | (3 << 22)     // pad_interval: every 16 DWORDs (64B)
        | (3 << 25);    // pad_amount: 4 DWORDs (16B)  -> 80B LDS row stride
  g1[1] = (tensor_w & 0xFFFF) << 16;                              // dim0[15:0]
  g1[2] = ((tensor_w >> 16) & 0xFFFF) | ((tensor_h & 0xFFFF) << 16); // dim0[31:16], dim1[15:0]
  g1[3] = ((tensor_h >> 16) & 0xFFFF) | ((tile_w & 0xFFFF) << 16);   // dim1[31:16], tile_dim0
  g1[4] = (tile_h & 0xFFFF);                    // tile_dim1; tile_dim2 = 0
  g1[5] = row_stride;                           // tensor_dim0_stride[31:0] (elems)
  g1[6] = 0;                                    // stride[47:32]=0, dim1_stride=0
  g1[7] = 0;
  i32x4 z4 = {0, 0, 0, 0};
#if __clang_major__ >= 23
  i32x8 z8 = {0, 0, 0, 0, 0, 0, 0, 0};
  __builtin_amdgcn_tensor_load_to_lds(g0, g1, z4, z4, z8, 0);
#else
  __builtin_amdgcn_tensor_load_to_lds(g0, g1, z4, z4, 0);
#endif
}

// ---------------------------------------------------------------- fp32 -> bf16
__global__ void k_cvt_bf16(const float* __restrict__ src, __bf16* __restrict__ dst, int n) {
  int i = blockIdx.x * blockDim.x + threadIdx.x;
  if (i < n) dst[i] = (__bf16)src[i];
}

// ------------------------------------- fp32 [K][N] -> bf16 [N][K] (transposed)
__global__ void k_cvtT_bf16(const float* __restrict__ src, __bf16* __restrict__ dst,
                            int Kdim, int Ndim) {
  int i = blockIdx.x * blockDim.x + threadIdx.x;   // over N*K, write-contiguous
  if (i < Kdim * Ndim) {
    int n = i / Kdim, k = i - n * Kdim;
    dst[i] = (__bf16)src[(size_t)k * Ndim + n];
  }
}

// ---- 64x64 block tile, 4 waves x (16M x 64N), double-buffered async LDS stage
// A: [M,K] row-major activations; Wt: [N,K] row-major (pre-transposed weights).
static __device__ __forceinline__ void gemm_tile64(
    const __bf16* __restrict__ A, const __bf16* __restrict__ Wt,
    int K, int bm0, int bn0, __bf16* sA, __bf16* sB, f32x8* acc) {
  const int t = threadIdx.x;
  const int lane = t & 31;
  const int wave = t >> 5;
  const int q0 = t * 2;               // 2 of 256 16B-chunks per thread per tile
  auto stage = [&](int k0, __bf16* dA, __bf16* dB) {
#pragma unroll
    for (int i = 0; i < 2; ++i) {
      int q = q0 + i;
      int row = q >> 2, seg = (q & 3) * 8;
      async_cp16(A  + (size_t)(bm0 + row) * K + k0 + seg, dA + row * LDT + seg);
      async_cp16(Wt + (size_t)(bn0 + row) * K + k0 + seg, dB + row * LDT + seg);
    }
  };
  stage(0, sA, sB);
  const int nk = K / KT;
  for (int kt = 0; kt < nk; ++kt) {
    __bf16* cA = sA + (kt & 1) * (64 * LDT);
    __bf16* cB = sB + (kt & 1) * (64 * LDT);
    async_wait0();                    // my copies (incl. current tile) done
    __syncthreads();                  // everyone's current tile resident
    if (kt + 1 < nk)                  // prefetch next tile into other buffer
      stage((kt + 1) * KT, sA + ((kt + 1) & 1) * (64 * LDT),
                           sB + ((kt + 1) & 1) * (64 * LDT));
    bf16x16 a;                        // wave's 16 M-rows, contiguous 32B per lane
    const __bf16* ar = cA + (wave * 16 + (lane & 15)) * LDT + ((lane >> 4) * 16);
#pragma unroll
    for (int e = 0; e < 16; ++e)
      a[e] = ar[e];
#pragma unroll
    for (int nt = 0; nt < 4; ++nt) {  // 4 N-tiles share one A-fragment
      bf16x16 bfr;                    // contiguous 32B per lane -> 2x ds_load_b128
      const __bf16* br = cB + (nt * 16 + (lane & 15)) * LDT + ((lane >> 4) * 16);
#pragma unroll
      for (int e = 0; e < 16; ++e)
        bfr[e] = br[e];
      acc[nt] = __builtin_amdgcn_wmma_f32_16x16x32_bf16(false, a, false, bfr,
                                                        (short)0, acc[nt], false, false);
    }
    __syncthreads();                  // all reads of cur buffer done before reuse
  }
}

// ------------------------------------------------- GEMM1: proj = x @ W_proj + b
// Epilogue: U -> silu(U) f32; V -> bf16 [B,H,hd,L] (transposed, vector stores);
//           Q/K -> bf16 [B,H,L,hd].
__global__ __launch_bounds__(128) void k_gemm_proj(
    const __bf16* __restrict__ A, const __bf16* __restrict__ Wt,
    const float* __restrict__ bias,
    float* __restrict__ UG, __bf16* __restrict__ VT,
    __bf16* __restrict__ QB, __bf16* __restrict__ KB) {
  __shared__ __bf16 sA[2 * 64 * LDT], sB[2 * 64 * LDT];
  const int lane = threadIdx.x & 31;
  const int wave = threadIdx.x >> 5;
  f32x8 acc[4] = {};
  gemm_tile64(A, Wt, D_, blockIdx.y * 64, blockIdx.x * 64, sA, sB, acc);

  const int bm = blockIdx.y * 64 + wave * 16;
  const int mBase = bm + ((lane >> 4) * 8);     // 8 consecutive M rows
  const int bi = mBase >> 11;                   // batch (m / L_)
  const int li0 = mBase & (L_ - 1);
#pragma unroll
  for (int nt = 0; nt < 4; ++nt) {
    const int col = blockIdx.x * 64 + nt * 16 + (lane & 15);
    const int region = col >> 10;     // 0=U 1=V 2=Q 3=K
    const int cin = col & 1023;
    const int h = cin >> 6, dd = cin & 63;
    const float bv = bias[col];
    if (region == 0) {
#pragma unroll
      for (int v = 0; v < 8; ++v) {
        float val = acc[nt][v] + bv;
        UG[(size_t)(mBase + v) * D_ + cin] = val * fast_sigmoid(val);
      }
    } else {
      bf16x8 pk;
#pragma unroll
      for (int v = 0; v < 8; ++v)
        pk[v] = (__bf16)(acc[nt][v] + bv);
      if (region == 1) {              // V transposed: L-contiguous -> b128 store
        *(bf16x8*)&VT[(((size_t)bi * H_ + h) * HD_ + dd) * L_ + li0] = pk;
      } else {
        __bf16* dst = (region == 2 ? QB : KB) +
                      (((size_t)bi * H_ + h) * L_ + li0) * HD_ + dd;
#pragma unroll
        for (int v = 0; v < 8; ++v) dst[(size_t)v * HD_] = pk[v];
      }
    }
  }
}

// ------------------------------------------------------------- in-place RoPE
__global__ void k_rope(__bf16* __restrict__ QB, __bf16* __restrict__ KB) {
  int idx = blockIdx.x * blockDim.x + threadIdx.x;  // B*H*L*32 threads exactly
  int d  = idx & 31;
  int l  = (idx >> 5) & (L_ - 1);
  int bh = idx >> 16;
  size_t base = ((size_t)bh * L_ + l) * HD_;
  float invf = __expf(-(float)(2 * d) * (9.2103403720f / 64.0f));  // 10000^(-2d/64)
  float th = (float)l * invf;
  float s, c;
  __sincosf(th, &s, &c);
  float q0 = (float)QB[base + d], q1 = (float)QB[base + d + 32];
  QB[base + d]      = (__bf16)(q0 * c - q1 * s);
  QB[base + d + 32] = (__bf16)(q1 * c + q0 * s);
  float k0 = (float)KB[base + d], k1 = (float)KB[base + d + 32];
  KB[base + d]      = (__bf16)(k0 * c - k1 * s);
  KB[base + d + 32] = (__bf16)(k1 * c + k0 * s);
}

// --------------------------- causal sigmoid attention, 1 wave per 16-row Q tile
// V is [B,H,hd,L] (transposed); each wave DMAs its V^T tile via the TDM.
__global__ __launch_bounds__(128) void k_attn(
    const __bf16* __restrict__ QB, const __bf16* __restrict__ KB,
    const __bf16* __restrict__ VT, float* __restrict__ AO) {
  __shared__ __bf16 sP[4][16 * 32];    // per-wave P tile (16 rows x 32 keys)
  __shared__ __bf16 sV[4][64 * LDT];   // per-wave V^T tile (64 dims x 32 keys, padded)
  const int lane = threadIdx.x & 31;
  const int wave = threadIdx.x >> 5;
  const int h = blockIdx.y, b = blockIdx.z;
  const int bh = b * H_ + h;
  const int qbase = (blockIdx.x * 4 + wave) * 16;
  const __bf16* qmat  = QB + (size_t)bh * L_ * HD_;
  const __bf16* kmat  = KB + (size_t)bh * L_ * HD_;
  const __bf16* vmatT = VT + (size_t)bh * HD_ * L_;

  // Q A-fragments (hd=64 -> two K=32 fragments), reused for all key tiles
  bf16x16 aq0, aq1;
  {
    const __bf16* qrow = qmat + (size_t)(qbase + (lane & 15)) * HD_;
    int kbf = (lane >> 4) * 16;
#pragma unroll
    for (int e = 0; e < 16; ++e) {
      int kk = kbf + e;
      aq0[e] = qrow[kk];
      aq1[e] = qrow[kk + 32];
    }
  }
  f32x8 o0 = {}, o1 = {}, o2 = {}, o3 = {};
  const int kend = qbase + 16;        // causal bound
  for (int kb = 0; kb < kend; kb += 32) {
    // ---- TDM: DMA this wave's 64x32 V^T tile into padded LDS (overlaps S phase)
    ds_wait0();                       // prior iter's ds reads of sV done
    tdm_load_tile_2d(vmatT + kb, &sV[wave][0],
                     /*tile_w=*/32, /*tile_h=*/HD_,
                     /*tensor_w=*/L_ - kb, /*tensor_h=*/HD_, /*row_stride=*/L_);
    // ---- S = Q K^T for 32 keys (two 16-key halves), sigmoid*mask -> LDS
#pragma unroll
    for (int hh = 0; hh < 2; ++hh) {
      int key = kb + hh * 16 + (lane & 15);
      const __bf16* krow = kmat + (size_t)key * HD_;
      bf16x16 bkA, bkB;
      int kk0 = (lane >> 4) * 16;
#pragma unroll
      for (int e = 0; e < 16; ++e) {  // contiguous 32B per lane -> b128 loads
        bkA[e] = krow[kk0 + e];
        bkB[e] = krow[kk0 + e + 32];
      }
      f32x8 s = {};
      s = __builtin_amdgcn_wmma_f32_16x16x32_bf16(false, aq0, false, bkA, (short)0, s, false, false);
      s = __builtin_amdgcn_wmma_f32_16x16x32_bf16(false, aq1, false, bkB, (short)0, s, false, false);
#pragma unroll
      for (int v = 0; v < 8; ++v) {
        int mrow = v + ((lane >> 4) * 8);
        int kg = kb + hh * 16 + (lane & 15);
        float p = fast_sigmoid(s[v] * 0.125f);       // scale = hd^-0.5
        if (kg > qbase + mrow) p = 0.0f;             // causal mask
        sP[wave][mrow * 32 + hh * 16 + (lane & 15)] = (__bf16)p;
      }
    }
    ds_wait0();                       // sP visible to this wave
    tensor_wait0();                   // V tile DMA complete (TENSORcnt)
    // ---- reload P as A-fragment (contiguous 32B per lane)
    bf16x16 pf;
    {
      int m = lane & 15;
      int kbf = (lane >> 4) * 16;
#pragma unroll
      for (int e = 0; e < 16; ++e)
        pf[e] = sP[wave][m * 32 + kbf + e];
    }
    // ---- O += P(16x32) x V(32x64), key-contiguous fragments from LDS
#pragma unroll
    for (int nt = 0; nt < 4; ++nt) {
      bf16x16 bv;
      const __bf16* vr = &sV[wave][(nt * 16 + (lane & 15)) * LDT + ((lane >> 4) * 16)];
#pragma unroll
      for (int e = 0; e < 16; ++e)
        bv[e] = vr[e];
      f32x8& oo = (nt == 0 ? o0 : nt == 1 ? o1 : nt == 2 ? o2 : o3);
      oo = __builtin_amdgcn_wmma_f32_16x16x32_bf16(false, pf, false, bv, (short)0, oo, false, false);
    }
  }
  // write attn_out back in [B, L, D] (heads interleaved)
#pragma unroll
  for (int nt = 0; nt < 4; ++nt) {
    const f32x8& oo = (nt == 0 ? o0 : nt == 1 ? o1 : nt == 2 ? o2 : o3);
#pragma unroll
    for (int v = 0; v < 8; ++v) {
      int mrow = v + ((lane >> 4) * 8);
      int dimc = nt * 16 + (lane & 15);
      AO[((size_t)b * L_ + qbase + mrow) * D_ + h * HD_ + dimc] = oo[v];
    }
  }
}

// ------------------------------------------- LayerNorm + silu(U) gate -> bf16
__global__ __launch_bounds__(256) void k_lngate(
    const float* __restrict__ AO, const float* __restrict__ UG,
    const float* __restrict__ gamma, const float* __restrict__ beta,
    __bf16* __restrict__ GT) {
  __shared__ float red[256];
  const int row = blockIdx.x, t = threadIdx.x;
  const float* a = AO + (size_t)row * D_;
  float vals[4], s = 0.f, ss = 0.f;
#pragma unroll
  for (int i = 0; i < 4; ++i) {
    float v = a[t + i * 256];
    vals[i] = v; s += v; ss += v * v;
  }
  red[t] = s; __syncthreads();
  for (int off = 128; off > 0; off >>= 1) {
    if (t < off) red[t] += red[t + off];
    __syncthreads();
  }
  float mu = red[0] * (1.0f / D_);
  __syncthreads();
  red[t] = ss; __syncthreads();
  for (int off = 128; off > 0; off >>= 1) {
    if (t < off) red[t] += red[t + off];
    __syncthreads();
  }
  float var = red[0] * (1.0f / D_) - mu * mu;
  float inv = rsqrtf(var + 1e-8f);
#pragma unroll
  for (int i = 0; i < 4; ++i) {
    int d = t + i * 256;
    float ln = (vals[i] - mu) * inv * gamma[d] + beta[d];
    GT[(size_t)row * D_ + d] = (__bf16)(UG[(size_t)row * D_ + d] * ln);
  }
}

// ----------------------------------- GEMM2: out = gated @ W_out + b_out + x
__global__ __launch_bounds__(128) void k_gemm_out(
    const __bf16* __restrict__ A, const __bf16* __restrict__ Wt,
    const float* __restrict__ bias, const float* __restrict__ xres,
    float* __restrict__ out) {
  __shared__ __bf16 sA[2 * 64 * LDT], sB[2 * 64 * LDT];
  const int lane = threadIdx.x & 31;
  const int wave = threadIdx.x >> 5;
  f32x8 acc[4] = {};
  gemm_tile64(A, Wt, D_, blockIdx.y * 64, blockIdx.x * 64, sA, sB, acc);

  const int bm = blockIdx.y * 64 + wave * 16;
#pragma unroll
  for (int nt = 0; nt < 4; ++nt) {
    const int n = blockIdx.x * 64 + nt * 16 + (lane & 15);
    const float bo = bias[n];
#pragma unroll
    for (int v = 0; v < 8; ++v) {
      int m = bm + v + ((lane >> 4) * 8);
      size_t idx = (size_t)m * D_ + n;
      out[idx] = acc[nt][v] + bo + xres[idx];
    }
  }
}

// -----------------------------------------------------------------------------
extern "C" void kernel_launch(void* const* d_in, const int* in_sizes, int n_in,
                              void* d_out, int out_size, void* d_ws, size_t ws_size,
                              hipStream_t stream) {
  const float* x      = (const float*)d_in[0];
  // d_in[1] = attn_mask: causal tril, handled analytically in-kernel
  const float* W_proj = (const float*)d_in[2];
  const float* b_proj = (const float*)d_in[3];
  const float* ln_g   = (const float*)d_in[4];
  const float* ln_b   = (const float*)d_in[5];
  const float* W_out  = (const float*)d_in[6];
  const float* b_out  = (const float*)d_in[7];
  float* out = (float*)d_out;

  char* ws = (char*)d_ws;
  auto take = [&](size_t bytes) { char* p = ws; ws += bytes; return p; };
  __bf16* XB  = (__bf16*)take((size_t)M_ * D_ * 2);    // x bf16
  __bf16* WPT = (__bf16*)take((size_t)D_ * N1_ * 2);   // W_proj^T bf16 [4096][1024]
  __bf16* WOT = (__bf16*)take((size_t)D_ * D_ * 2);    // W_out^T bf16 [1024][1024]
  float*  UG  = (float*) take((size_t)M_ * D_ * 4);    // silu(U) f32
  __bf16* VT  = (__bf16*)take((size_t)M_ * D_ * 2);    // V^T [B,H,hd,L]
  __bf16* QB  = (__bf16*)take((size_t)M_ * D_ * 2);    // Q  [B,H,L,hd]
  __bf16* KB  = (__bf16*)take((size_t)M_ * D_ * 2);    // K  [B,H,L,hd]
  float*  AO  = (float*) take((size_t)M_ * D_ * 4);    // attn_out f32
  __bf16* GT  = (__bf16*)take((size_t)M_ * D_ * 2);    // gated bf16

  int n;
  n = M_ * D_;   k_cvt_bf16 <<<(n + 255) / 256, 256, 0, stream>>>(x, XB, n);
  n = D_ * N1_;  k_cvtT_bf16<<<(n + 255) / 256, 256, 0, stream>>>(W_proj, WPT, D_, N1_);
  n = D_ * D_;   k_cvtT_bf16<<<(n + 255) / 256, 256, 0, stream>>>(W_out, WOT, D_, D_);

  k_gemm_proj<<<dim3(N1_ / 64, M_ / 64), 128, 0, stream>>>(XB, WPT, b_proj, UG, VT, QB, KB);
  k_rope<<<(B_ * H_ * L_ * 32) / 256, 256, 0, stream>>>(QB, KB);
  k_attn<<<dim3(L_ / 64, H_, B_), 128, 0, stream>>>(QB, KB, VT, AO);
  k_lngate<<<M_, 256, 0, stream>>>(AO, UG, ln_g, ln_b, GT);
  k_gemm_out<<<dim3(D_ / 64, M_ / 64), 128, 0, stream>>>(GT, WOT, b_out, x, out);
}
